// CLIP_69458211111588
// MI455X (gfx1250) — matrix operand order repaired and checked by reading, compiled, not run on hardware
//
#include <hip/hip_runtime.h>
#include <hip/hip_bf16.h>

// ---------------------------------------------------------------------------
// Problem constants (from the reference)
// ---------------------------------------------------------------------------
#define BB  32
#define TT  512
#define FF  1024
#define DM  256
#define DFF 512
#define LN_EPS 1e-5f

// Use gfx1250 async global->LDS staging in the GEMM (set 0 to fall back to
// the global_load_b128 + ds_store_b128 path).
#define CLIP_USE_ASYNC 1

// ---------------------------------------------------------------------------
// Types for CDNA5 WMMA
// ---------------------------------------------------------------------------
typedef __bf16 v16bf __attribute__((ext_vector_type(16)));
typedef float  v8f   __attribute__((ext_vector_type(8)));
typedef unsigned int u32x4 __attribute__((ext_vector_type(4)));

__device__ __forceinline__ unsigned short f2bf(float f) {
  union { float f; unsigned u; } x; x.f = f;
  unsigned r = x.u + 0x7FFFu + ((x.u >> 16) & 1u);   // round-to-nearest-even
  return (unsigned short)(r >> 16);
}
__device__ __forceinline__ float bf2f(unsigned short h) {
  union { unsigned u; float f; } x; x.u = ((unsigned)h) << 16; return x.f;
}
__device__ __forceinline__ unsigned short to_bf(float x)          { return f2bf(x); }
__device__ __forceinline__ unsigned short to_bf(unsigned short x) { return x; }

// ---------------------------------------------------------------------------
// Generic batched GEMM:  C[M,N] (f32 acc) = A[M,K](bf16, row-major)
//                                         @ B,  fed as Bt[N,K](bf16, row-major)
// Block tile 128x128, K-step 32, 8 waves (2x4), wave tile 64x32 (4x2 WMMA frags)
// Double-buffered LDS, async global->LDS staging, one barrier per K-step.
// EPI: 0 f32 | 1 bf16 | 2 bf16(+bias) | 3 f32+bf16 | 4 f32 sigmoid(+bias)
//      5 bf16 relu(+bias)
// ---------------------------------------------------------------------------
#define GBM 128
#define GBN 128
#define GBK 32
#define LSTR 48   // LDS row stride in bf16 elems: 96B, 16B-aligned, pad vs banks

template<int EPI>
__global__ __launch_bounds__(256)
void clip_gemm_bf16(const unsigned short* __restrict__ A,  long long bsA,
                    const unsigned short* __restrict__ Bt, long long bsB,
                    float* __restrict__ Cf, unsigned short* __restrict__ Cbf,
                    long long bsC, const float* __restrict__ bias,
                    int M, int N, int K) {
  __shared__ __attribute__((aligned(16))) unsigned short smA[2][GBM * LSTR];
  __shared__ __attribute__((aligned(16))) unsigned short smB[2][GBN * LSTR];

  const int b = blockIdx.z;
  A  += (long long)b * bsA;
  Bt += (long long)b * bsB;
  const long long cOff = (long long)b * bsC;

  const int m0 = blockIdx.y * GBM;
  const int n0 = blockIdx.x * GBN;
  const int tid  = threadIdx.x;
  const int lane = tid & 31;
  const int wid  = tid >> 5;
  const int wm   = (wid >> 2) * 64;   // 0 / 64
  const int wn   = (wid & 3) * 32;    // 0 / 32 / 64 / 96
  const int l16  = lane & 15;
  const int lh   = lane >> 4;         // half-wave select

  v8f acc[4][2];
#pragma unroll
  for (int mi = 0; mi < 4; ++mi)
#pragma unroll
    for (int ni = 0; ni < 2; ++ni)
#pragma unroll
      for (int j = 0; j < 8; ++j) acc[mi][ni][j] = 0.f;

  const int lrow = tid >> 1;          // 0..127
  const int lcol = (tid & 1) * 16;    // 0 / 16

  // stage K-tile k0 into LDS buffer p (each thread moves 32B of A + 32B of Bt)
  auto stage = [&](int k0, int p) {
    const unsigned short* ga = A  + (long long)(m0 + lrow) * K + k0 + lcol;
    const unsigned short* gb = Bt + (long long)(n0 + lrow) * K + k0 + lcol;
#if CLIP_USE_ASYNC
    const unsigned la = (unsigned)(unsigned long long)&smA[p][lrow * LSTR + lcol];
    const unsigned lb = (unsigned)(unsigned long long)&smB[p][lrow * LSTR + lcol];
    // ASYNCcnt-tracked DMA: 16B global->LDS copy, offset added to both addresses
    asm volatile("global_load_async_to_lds_b128 %0, %1, off"            :: "v"(la), "v"(ga) : "memory");
    asm volatile("global_load_async_to_lds_b128 %0, %1, off offset:16"  :: "v"(la), "v"(ga) : "memory");
    asm volatile("global_load_async_to_lds_b128 %0, %1, off"            :: "v"(lb), "v"(gb) : "memory");
    asm volatile("global_load_async_to_lds_b128 %0, %1, off offset:16"  :: "v"(lb), "v"(gb) : "memory");
#else
    *reinterpret_cast<u32x4*>(&smA[p][lrow * LSTR + lcol])     = *reinterpret_cast<const u32x4*>(ga);
    *reinterpret_cast<u32x4*>(&smA[p][lrow * LSTR + lcol + 8]) = *reinterpret_cast<const u32x4*>(ga + 8);
    *reinterpret_cast<u32x4*>(&smB[p][lrow * LSTR + lcol])     = *reinterpret_cast<const u32x4*>(gb);
    *reinterpret_cast<u32x4*>(&smB[p][lrow * LSTR + lcol + 8]) = *reinterpret_cast<const u32x4*>(gb + 8);
#endif
  };

  const int nk = K / GBK;
  stage(0, 0);

  for (int k = 0; k < nk; ++k) {
    const int p = k & 1;
#if CLIP_USE_ASYNC
    asm volatile("s_wait_asynccnt 0" ::: "memory");   // own tile DMA complete
    __syncthreads();                                  // everyone's tile in LDS[p]
    if (k + 1 < nk) stage((k + 1) * GBK, p ^ 1);      // prefetch next tile
#else
    __syncthreads();                                  // LDS[p] writes visible
    if (k + 1 < nk) { /* staged below to keep two-barrier scheme */ }
#endif

    union Frag { v16bf v; u32x4 q[2]; };
    Frag fa[4], fb[2];
    // A fragment: lane holds row M=l16, K chunks [8*lh, 8*lh+7] and +16
#pragma unroll
    for (int mi = 0; mi < 4; ++mi) {
      const int base = (wm + mi * 16 + l16) * LSTR;
      fa[mi].q[0] = *reinterpret_cast<const u32x4*>(&smA[p][base + lh * 8]);
      fa[mi].q[1] = *reinterpret_cast<const u32x4*>(&smA[p][base + 16 + lh * 8]);
    }
    // B fragment: lane holds column N=l16, K chunks [8*lh, 8*lh+7] and +16
#pragma unroll
    for (int ni = 0; ni < 2; ++ni) {
      const int base = (wn + ni * 16 + l16) * LSTR;
      fb[ni].q[0] = *reinterpret_cast<const u32x4*>(&smB[p][base + lh * 8]);
      fb[ni].q[1] = *reinterpret_cast<const u32x4*>(&smB[p][base + 16 + lh * 8]);
    }
#pragma unroll
    for (int mi = 0; mi < 4; ++mi)
#pragma unroll
      for (int ni = 0; ni < 2; ++ni)
        acc[mi][ni] = __builtin_amdgcn_wmma_f32_16x16x32_bf16(
            false, fa[mi].v, false, fb[ni].v, (short)0, acc[mi][ni],
            false, false);

#if !CLIP_USE_ASYNC
    __syncthreads();
    if (k + 1 < nk) stage((k + 1) * GBK, p ^ 1);
#endif
  }

  // Epilogue. C/D layout: VGPR j -> (M = 8*lh + j, N = l16)
#pragma unroll
  for (int mi = 0; mi < 4; ++mi) {
#pragma unroll
    for (int ni = 0; ni < 2; ++ni) {
      const int n  = n0 + wn + ni * 16 + l16;
      const int mB = m0 + wm + mi * 16 + lh * 8;
      float bv = 0.f;
      if constexpr (EPI == 2 || EPI == 4 || EPI == 5) bv = bias[n];
#pragma unroll
      for (int j = 0; j < 8; ++j) {
        float v = acc[mi][ni][j];
        const long long idx = cOff + (long long)(mB + j) * N + n;
        if constexpr (EPI == 0) {
          Cf[idx] = v;
        } else if constexpr (EPI == 1) {
          Cbf[idx] = f2bf(v);
        } else if constexpr (EPI == 2) {
          Cbf[idx] = f2bf(v + bv);
        } else if constexpr (EPI == 3) {
          Cf[idx] = v; Cbf[idx] = f2bf(v);
        } else if constexpr (EPI == 4) {
          Cf[idx] = 1.f / (1.f + __expf(-(v + bv)));
        } else {
          Cbf[idx] = f2bf(fmaxf(v + bv, 0.f));
        }
      }
    }
  }
}

// ---------------------------------------------------------------------------
// Tiled transpose (+ f32->bf16 convert).  dst[c*R + r] = cvt(src[r*C + c]).
// R, C multiples of 32. Batched via grid.z.
// ---------------------------------------------------------------------------
template<typename T>
__global__ void clip_transpose_cvt(const T* __restrict__ src,
                                   unsigned short* __restrict__ dst,
                                   int R, int C, long long bsSrc, long long bsDst) {
  __shared__ unsigned short tile[32][33];
  src += (long long)blockIdx.z * bsSrc;
  dst += (long long)blockIdx.z * bsDst;
  const int c0 = blockIdx.x * 32, r0 = blockIdx.y * 32;
  const int tx = threadIdx.x, ty = threadIdx.y;
#pragma unroll
  for (int i = 0; i < 4; ++i) {
    const int r = r0 + ty + i * 8;
    tile[ty + i * 8][tx] = to_bf(src[(long long)r * C + c0 + tx]);
  }
  __syncthreads();
#pragma unroll
  for (int i = 0; i < 4; ++i) {
    const int c = c0 + ty + i * 8;
    dst[(long long)c * R + r0 + tx] = tile[tx][ty + i * 8];
  }
}

// ---------------------------------------------------------------------------
// aging reductions: svec[b][d] = sum_t aging[b][t]*W_dm[t][d];
//                   colsum[d]  = sum_t W_dm[t][d]
// ---------------------------------------------------------------------------
__global__ void clip_aging_reduce(const float* __restrict__ aging,
                                  const float* __restrict__ W_dm,
                                  float* __restrict__ svec,
                                  float* __restrict__ colsum) {
  const int d = threadIdx.x;       // 256 threads = DM
  float sv[BB];
#pragma unroll
  for (int b = 0; b < BB; ++b) sv[b] = 0.f;
  float cs = 0.f;
  for (int t = 0; t < TT; ++t) {
    const float w = W_dm[t * DM + d];
    cs += w;
#pragma unroll
    for (int b = 0; b < BB; ++b) sv[b] += aging[b * TT + t] * w;
  }
  colsum[d] = cs;
#pragma unroll
  for (int b = 0; b < BB; ++b) svec[b * DM + d] = sv[b];
}

// ---------------------------------------------------------------------------
// LayerNorm helpers (one block = one (b,f) row of DM=256 elements)
// ---------------------------------------------------------------------------
__device__ __forceinline__ float block_reduce256(float v, float* red) {
  const int t = threadIdx.x;
  red[t] = v; __syncthreads();
  for (int s = 128; s > 0; s >>= 1) {
    if (t < s) red[t] += red[t + s];
    __syncthreads();
  }
  const float r = red[0]; __syncthreads();
  return r;
}

// af (analytic rank-1 form) + LN -> bf16
__global__ void clip_af_ln(const float* __restrict__ W_age,
                           const float* __restrict__ b_age,
                           const float* __restrict__ b_dm,
                           const float* __restrict__ svec,
                           const float* __restrict__ colsum,
                           const float* __restrict__ ln_g,
                           const float* __restrict__ ln_b,
                           unsigned short* __restrict__ af_bf) {
  __shared__ float red[256];
  const int f = blockIdx.x, b = blockIdx.y, d = threadIdx.x;
  const float raw = W_age[f] * svec[b * DM + d] + b_age[f] * colsum[d] + b_dm[d];
  const float mean = block_reduce256(raw, red) * (1.f / DM);
  const float c = raw - mean;
  const float var = block_reduce256(c * c, red) * (1.f / DM);
  const float y = c * rsqrtf(var + LN_EPS) * ln_g[d] + ln_b[d];
  af_bf[((long long)(b * FF + f)) * DM + d] = f2bf(y);
}

// sf = LN(sf_raw + b_enc) -> bf16
__global__ void clip_sf_ln(const float* __restrict__ sf_raw,
                           const float* __restrict__ b_enc,
                           const float* __restrict__ ln_g,
                           const float* __restrict__ ln_b,
                           unsigned short* __restrict__ sf_bf) {
  __shared__ float red[256];
  const int f = blockIdx.x, b = blockIdx.y, d = threadIdx.x;
  const long long idx = ((long long)(b * FF + f)) * DM + d;
  const float raw = sf_raw[idx] + b_enc[d];
  const float mean = block_reduce256(raw, red) * (1.f / DM);
  const float c = raw - mean;
  const float var = block_reduce256(c * c, red) * (1.f / DM);
  sf_bf[idx] = f2bf(c * rsqrtf(var + LN_EPS) * ln_g[d] + ln_b[d]);
}

// ---------------------------------------------------------------------------
// Row softmax over F=1024 (block = one (b,n) row, 256 threads x 4)
// ---------------------------------------------------------------------------
__global__ void clip_softmax(const float* __restrict__ logits,
                             unsigned short* __restrict__ adj_bf) {
  __shared__ float red[256];
  const int n = blockIdx.x, b = blockIdx.y, t = threadIdx.x;
  const long long base = ((long long)(b * FF + n)) * FF;
  float v[4];
  float mx = -3.4e38f;
#pragma unroll
  for (int k = 0; k < 4; ++k) { v[k] = logits[base + t + k * 256]; mx = fmaxf(mx, v[k]); }
  red[t] = mx; __syncthreads();
  for (int s = 128; s > 0; s >>= 1) { if (t < s) red[t] = fmaxf(red[t], red[t + s]); __syncthreads(); }
  mx = red[0]; __syncthreads();
  float sum = 0.f;
#pragma unroll
  for (int k = 0; k < 4; ++k) { v[k] = __expf(v[k] - mx); sum += v[k]; }
  sum = block_reduce256(sum, red);
  const float inv = 1.f / sum;
#pragma unroll
  for (int k = 0; k < 4; ++k) adj_bf[base + t + k * 256] = f2bf(v[k] * inv);
}

// ---------------------------------------------------------------------------
// Loss: per-row cosine between z_s and z_a rows, then fixed-order reduce
// ---------------------------------------------------------------------------
__global__ void clip_loss_rows(const float* __restrict__ zs,
                               const float* __restrict__ za,
                               float* __restrict__ rowbuf) {
  __shared__ float red[256];
  const int n = blockIdx.x, b = blockIdx.y, t = threadIdx.x;
  const long long base = ((long long)(b * FF + n)) * FF;
  float dot = 0.f, ns2 = 0.f, na2 = 0.f;
#pragma unroll
  for (int k = 0; k < 4; ++k) {
    const float s = zs[base + t + k * 256];
    const float a = za[base + t + k * 256];
    dot += s * a; ns2 += s * s; na2 += a * a;
  }
  dot = block_reduce256(dot, red);
  ns2 = block_reduce256(ns2, red);
  na2 = block_reduce256(na2, red);
  if (t == 0) {
    const float dn = fmaxf(sqrtf(ns2), 1e-12f) * fmaxf(sqrtf(na2), 1e-12f);
    rowbuf[b * FF + n] = dot / dn;
  }
}

__global__ void clip_loss_reduce(const float* __restrict__ rowbuf,
                                 float* __restrict__ out) {
  __shared__ float red[256];
  const int t = threadIdx.x;
  float s = 0.f;
  for (int i = t; i < BB * FF; i += 256) s += rowbuf[i];
  s = block_reduce256(s, red);
  if (t == 0) out[0] = 0.1f * (1.f - s / (float)(BB * FF));
}

// ---------------------------------------------------------------------------
// enc = z_s + gate_s*z_a + z_a + gate_a*z_s   (gates already sigmoided)
// ---------------------------------------------------------------------------
__global__ void clip_combine(const float* __restrict__ zs,
                             const float* __restrict__ za,
                             const float* __restrict__ gs,
                             const float* __restrict__ ga,
                             unsigned short* __restrict__ enc_bf,
                             long long total) {
  for (long long i = blockIdx.x * 256ll + threadIdx.x; i < total;
       i += (long long)gridDim.x * 256ll) {
    const float s = zs[i], a = za[i];
    enc_bf[i] = f2bf(s + gs[i] * a + a + ga[i] * s);
  }
}

// ---------------------------------------------------------------------------
// Decoder tail
// ---------------------------------------------------------------------------
// v[b][f] = relu(enc@W1+b1)[b,f,:] . W_dec2 + b_dec2   (first dec, col of 1)
__global__ void clip_dec2_rowdot(const unsigned short* __restrict__ d1,
                                 const float* __restrict__ W_dec2,
                                 const float* __restrict__ b_dec2,
                                 float* __restrict__ vbuf) {
  __shared__ float red[128];
  const int f = blockIdx.x, b = blockIdx.y, t = threadIdx.x;
  const long long base = ((long long)(b * FF + f)) * DFF;
  float s = 0.f;
#pragma unroll
  for (int k = 0; k < 4; ++k) {
    const int j = t + k * 128;
    s += bf2f(d1[base + j]) * W_dec2[j];
  }
  red[t] = s; __syncthreads();
  for (int st = 64; st > 0; st >>= 1) { if (t < st) red[t] += red[t + st]; __syncthreads(); }
  if (t == 0) vbuf[b * FF + f] = red[0] + b_dec2[0];
}

// second dec: per-batch (B,1,F)@W1 -> relu -> .W_dec2 -> |.|
__global__ void clip_dec_final(const float* __restrict__ vbuf,
                               const float* __restrict__ W_dec1,
                               const float* __restrict__ b_dec1,
                               const float* __restrict__ W_dec2,
                               const float* __restrict__ b_dec2,
                               float* __restrict__ out) {
  __shared__ float red[512];
  const int b = blockIdx.x, j = threadIdx.x;   // 512 threads
  float h = b_dec1[j];
  for (int f = 0; f < FF; ++f) h += vbuf[b * FF + f] * W_dec1[f * DFF + j];
  h = fmaxf(h, 0.f);
  red[j] = h * W_dec2[j]; __syncthreads();
  for (int s = 256; s > 0; s >>= 1) { if (j < s) red[j] += red[j + s]; __syncthreads(); }
  if (j == 0) out[1 + b] = fabsf(red[0] + b_dec2[0]);
}

// ---------------------------------------------------------------------------
// Host-side orchestration
// ---------------------------------------------------------------------------
extern "C" void kernel_launch(void* const* d_in, const int* in_sizes, int n_in,
                              void* d_out, int out_size, void* d_ws, size_t ws_size,
                              hipStream_t stream) {
  const float* x_enc   = (const float*)d_in[0];
  const float* aging   = (const float*)d_in[1];
  const float* W_enc   = (const float*)d_in[2];
  const float* b_enc   = (const float*)d_in[3];
  const float* W_age   = (const float*)d_in[4];
  const float* b_age   = (const float*)d_in[5];
  const float* W_dm    = (const float*)d_in[6];
  const float* b_dm    = (const float*)d_in[7];
  const float* ln_g    = (const float*)d_in[8];
  const float* ln_b    = (const float*)d_in[9];
  const float* W_gcn_s = (const float*)d_in[10];
  const float* b_gcn_s = (const float*)d_in[11];
  const float* W_gcn_a = (const float*)d_in[12];
  const float* b_gcn_a = (const float*)d_in[13];
  const float* W_gate_s= (const float*)d_in[14];
  const float* b_gate_s= (const float*)d_in[15];
  const float* W_gate_a= (const float*)d_in[16];
  const float* b_gate_a= (const float*)d_in[17];
  const float* W_dec1  = (const float*)d_in[18];
  const float* b_dec1  = (const float*)d_in[19];
  const float* W_dec2  = (const float*)d_in[20];
  const float* b_dec2  = (const float*)d_in[21];
  float* out = (float*)d_out;

  // -------- bump allocator over d_ws --------
  size_t off = 0;
  auto alloc = [&](size_t bytes) -> char* {
    off = (off + 255) & ~(size_t)255;
    char* p = (char*)d_ws + off;
    off += bytes;
    return p;
  };
  const size_t nBF  = (size_t)BB * FF;                 // 32768 rows
  const size_t nBFD = nBF * DM;                        // 8.4M
  const size_t nBFF = nBF * FF;                        // 33.5M

  unsigned short* xT_bf   = (unsigned short*)alloc((size_t)BB * FF * TT * 2); // also reused as D1
  float*          sf_raw  = (float*)alloc(nBFD * 4);
  unsigned short* sf_bf   = (unsigned short*)alloc(nBFD * 2);
  unsigned short* sfT_bf  = (unsigned short*)alloc(nBFD * 2);
  unsigned short* af_bf   = (unsigned short*)alloc(nBFD * 2);
  unsigned short* afT_bf  = (unsigned short*)alloc(nBFD * 2);
  float*          BUF1    = (float*)alloc(nBFF * 4);   // logits_s -> gate_s
  float*          BUF2    = (float*)alloc(nBFF * 4);   // logits_a -> gate_a
  unsigned short* ADJ_BF  = (unsigned short*)alloc(nBFF * 2); // adj, then enc
  unsigned short* T_BF    = (unsigned short*)alloc(nBFD * 2);
  unsigned short* HS_BF   = (unsigned short*)alloc(nBFD * 2);
  unsigned short* HA_BF   = (unsigned short*)alloc(nBFD * 2);
  float*          Z_S     = (float*)alloc(nBFF * 4);
  float*          Z_A     = (float*)alloc(nBFF * 4);
  unsigned short* ZS_BF   = (unsigned short*)alloc(nBFF * 2);
  unsigned short* ZA_BF   = (unsigned short*)alloc(nBFF * 2);
  float*          svec    = (float*)alloc((size_t)BB * DM * 4);
  float*          colsum  = (float*)alloc((size_t)DM * 4);
  float*          rowbuf  = (float*)alloc(nBF * 4);
  float*          vbuf    = (float*)alloc(nBF * 4);
  unsigned short* WencT   = (unsigned short*)alloc((size_t)DM * TT * 2);
  unsigned short* WgcnsT  = (unsigned short*)alloc((size_t)DM * DM * 2);
  unsigned short* WgcnaT  = (unsigned short*)alloc((size_t)DM * DM * 2);
  unsigned short* WgateTs = (unsigned short*)alloc((size_t)FF * FF * 2);
  unsigned short* WgateTa = (unsigned short*)alloc((size_t)FF * FF * 2);
  unsigned short* Wdec1T  = (unsigned short*)alloc((size_t)DFF * FF * 2);
  unsigned short* ENC_BF  = ADJ_BF;                    // reuse (adj consumed first)
  unsigned short* D1      = xT_bf;                     // reuse (x consumed first)

  const dim3 t328(32, 8);
  // ---- weight transposes (f32 -> bf16, Bt = N x K layout) ----
  clip_transpose_cvt<float><<<dim3(DM/32, TT/32, 1), t328, 0, stream>>>(W_enc, WencT, TT, DM, 0, 0);
  clip_transpose_cvt<float><<<dim3(DM/32, DM/32, 1), t328, 0, stream>>>(W_gcn_s, WgcnsT, DM, DM, 0, 0);
  clip_transpose_cvt<float><<<dim3(DM/32, DM/32, 1), t328, 0, stream>>>(W_gcn_a, WgcnaT, DM, DM, 0, 0);
  clip_transpose_cvt<float><<<dim3(FF/32, FF/32, 1), t328, 0, stream>>>(W_gate_s, WgateTs, FF, FF, 0, 0);
  clip_transpose_cvt<float><<<dim3(FF/32, FF/32, 1), t328, 0, stream>>>(W_gate_a, WgateTa, FF, FF, 0, 0);
  clip_transpose_cvt<float><<<dim3(DFF/32, FF/32, 1), t328, 0, stream>>>(W_dec1, Wdec1T, FF, DFF, 0, 0);
  // x_enc (B,T,F) -> xT_bf (B,F,T)
  clip_transpose_cvt<float><<<dim3(FF/32, TT/32, BB), t328, 0, stream>>>(
      x_enc, xT_bf, TT, FF, (long long)TT * FF, (long long)FF * TT);

  // ---- aging branch (analytic) ----
  clip_aging_reduce<<<1, 256, 0, stream>>>(aging, W_dm, svec, colsum);
  clip_af_ln<<<dim3(FF, BB), 256, 0, stream>>>(W_age, b_age, b_dm, svec, colsum, ln_g, ln_b, af_bf);

  // ---- sf = xT @ W_enc, then LN ----
  clip_gemm_bf16<0><<<dim3(DM/GBN, FF/GBM, BB), 256, 0, stream>>>(
      xT_bf, (long long)FF * TT, WencT, 0, sf_raw, nullptr, (long long)FF * DM,
      nullptr, FF, DM, TT);
  clip_sf_ln<<<dim3(FF, BB), 256, 0, stream>>>(sf_raw, b_enc, ln_g, ln_b, sf_bf);

  // ---- transposed bf16 copies (Bt operands for adj@x) ----
  clip_transpose_cvt<unsigned short><<<dim3(DM/32, FF/32, BB), t328, 0, stream>>>(
      sf_bf, sfT_bf, FF, DM, (long long)FF * DM, (long long)DM * FF);
  clip_transpose_cvt<unsigned short><<<dim3(DM/32, FF/32, BB), t328, 0, stream>>>(
      af_bf, afT_bf, FF, DM, (long long)FF * DM, (long long)DM * FF);

  // ---- adjacency logits (X @ X^T : Bt = X itself) ----
  clip_gemm_bf16<0><<<dim3(FF/GBN, FF/GBM, BB), 256, 0, stream>>>(
      sf_bf, (long long)FF * DM, sf_bf, (long long)FF * DM, BUF1, nullptr,
      (long long)FF * FF, nullptr, FF, FF, DM);
  clip_gemm_bf16<0><<<dim3(FF/GBN, FF/GBM, BB), 256, 0, stream>>>(
      af_bf, (long long)FF * DM, af_bf, (long long)FF * DM, BUF2, nullptr,
      (long long)FF * FF, nullptr, FF, FF, DM);

  // ---- view s: softmax -> adj@sf -> @W_gcn_s + b ----
  clip_softmax<<<dim3(FF, BB), 256, 0, stream>>>(BUF1, ADJ_BF);
  clip_gemm_bf16<1><<<dim3(DM/GBN, FF/GBM, BB), 256, 0, stream>>>(
      ADJ_BF, (long long)FF * FF, sfT_bf, (long long)DM * FF, nullptr, T_BF,
      (long long)FF * DM, nullptr, FF, DM, FF);
  clip_gemm_bf16<2><<<dim3(DM/GBN, FF/GBM, BB), 256, 0, stream>>>(
      T_BF, (long long)FF * DM, WgcnsT, 0, nullptr, HS_BF, (long long)FF * DM,
      b_gcn_s, FF, DM, DM);

  // ---- view a ----
  clip_softmax<<<dim3(FF, BB), 256, 0, stream>>>(BUF2, ADJ_BF);
  clip_gemm_bf16<1><<<dim3(DM/GBN, FF/GBM, BB), 256, 0, stream>>>(
      ADJ_BF, (long long)FF * FF, afT_bf, (long long)DM * FF, nullptr, T_BF,
      (long long)FF * DM, nullptr, FF, DM, FF);
  clip_gemm_bf16<2><<<dim3(DM/GBN, FF/GBM, BB), 256, 0, stream>>>(
      T_BF, (long long)FF * DM, WgcnaT, 0, nullptr, HA_BF, (long long)FF * DM,
      b_gcn_a, FF, DM, DM);

  // ---- z_s = h_s @ sf^T, z_a = h_a @ sf^T (Bt = sf) ----
  clip_gemm_bf16<3><<<dim3(FF/GBN, FF/GBM, BB), 256, 0, stream>>>(
      HS_BF, (long long)FF * DM, sf_bf, (long long)FF * DM, Z_S, ZS_BF,
      (long long)FF * FF, nullptr, FF, FF, DM);
  clip_gemm_bf16<3><<<dim3(FF/GBN, FF/GBM, BB), 256, 0, stream>>>(
      HA_BF, (long long)FF * DM, sf_bf, (long long)FF * DM, Z_A, ZA_BF,
      (long long)FF * FF, nullptr, FF, FF, DM);

  // ---- loss ----
  clip_loss_rows<<<dim3(FF, BB), 256, 0, stream>>>(Z_S, Z_A, rowbuf);
  clip_loss_reduce<<<1, 256, 0, stream>>>(rowbuf, out);

  // ---- gates: sigmoid(z_a@Wg_s + b), sigmoid(z_s@Wg_a + b) ----
  clip_gemm_bf16<4><<<dim3(FF/GBN, FF/GBM, BB), 256, 0, stream>>>(
      ZA_BF, (long long)FF * FF, WgateTs, 0, BUF1, nullptr, (long long)FF * FF,
      b_gate_s, FF, FF, FF);
  clip_gemm_bf16<4><<<dim3(FF/GBN, FF/GBM, BB), 256, 0, stream>>>(
      ZS_BF, (long long)FF * FF, WgateTa, 0, BUF2, nullptr, (long long)FF * FF,
      b_gate_a, FF, FF, FF);

  // ---- fuse ----
  clip_combine<<<8192, 256, 0, stream>>>(Z_S, Z_A, BUF1, BUF2, ENC_BF,
                                         (long long)BB * FF * FF);

  // ---- decoder ----
  clip_gemm_bf16<5><<<dim3(DFF/GBN, FF/GBM, BB), 256, 0, stream>>>(
      ENC_BF, (long long)FF * FF, Wdec1T, 0, nullptr, D1, (long long)FF * DFF,
      b_dec1, FF, DFF, FF);
  clip_dec2_rowdot<<<dim3(FF, BB), 128, 0, stream>>>(D1, W_dec2, b_dec2, vbuf);
  clip_dec_final<<<BB, 512, 0, stream>>>(vbuf, W_dec1, b_dec1, W_dec2, b_dec2, out);
}
